// LongRangeAttention_63230508531949
// MI455X (gfx1250) — matrix operand order repaired and checked by reading, not compiled, not run on hardware
//
#include <hip/hip_runtime.h>
#include <hip/hip_bf16.h>

// ---------------------------------------------------------------------------
// CDNA5 (gfx1250) block-local attention
//   - all GEMMs via v_wmma_f32_16x16x32_bf16 (f32 accumulate)
//   - Q/K staged by Tensor Data Mover (tensor_load_to_lds + s_wait_tensorcnt)
//   - two-pass online softmax to keep attention accumulators spill-free
// ---------------------------------------------------------------------------

typedef __attribute__((ext_vector_type(16))) __bf16 bf16x16;
typedef __attribute__((ext_vector_type(8)))  __bf16 bf16x8;
typedef __attribute__((ext_vector_type(8)))  float  f32x8;
typedef __attribute__((ext_vector_type(4)))  unsigned int u32x4;
typedef __attribute__((ext_vector_type(4)))  int i32x4;
typedef __attribute__((ext_vector_type(8)))  int i32x8;

#define DEVINL __device__ __forceinline__

#if __has_builtin(__builtin_amdgcn_tensor_load_to_lds)
#define HAVE_TDM 1
#else
#define HAVE_TDM 0
#endif

DEVINL void sched_fence() {
#if __has_builtin(__builtin_amdgcn_sched_barrier)
  __builtin_amdgcn_sched_barrier(0);
#endif
}

DEVINL unsigned short f2bf(float f) {
  unsigned u = __float_as_uint(f);
  u += 0x7FFFu + ((u >> 16) & 1u);   // round to nearest even
  return (unsigned short)(u >> 16);
}

DEVINL float bf2f(unsigned short h) {
  return __uint_as_float(((unsigned)h) << 16);
}

DEVINL f32x8 zero8() {
  f32x8 z;
#pragma unroll
  for (int e = 0; e < 8; ++e) z[e] = 0.0f;
  return z;
}

DEVINL f32x8 wmma_bf16(bf16x16 a, bf16x16 b, f32x8 c) {
  return __builtin_amdgcn_wmma_f32_16x16x32_bf16(
      false, a, false, b, (short)0, c, false, false);
}

// A fragment (16x32 bf16) from row-major LDS tile (stride in halves).
DEVINL bf16x16 load_frag_A(const unsigned short* base, int stride, int k0, int lane) {
  const int r  = lane & 15;
  const int kb = (lane >> 4) * 8;
  const unsigned short* p = base + r * stride + k0 + kb;
  bf16x8 lo = *(const bf16x8*)p;
  bf16x8 hi = *(const bf16x8*)(p + 16);
  bf16x16 a;
#pragma unroll
  for (int i = 0; i < 8; ++i) { a[i] = lo[i]; a[i + 8] = hi[i]; }
  return a;
}

// B fragment (32x16 bf16) from N-major LDS (each row n holds K contiguous).
DEVINL bf16x16 load_frag_B(const unsigned short* base, int stride, int k0, int lane) {
  const int n  = lane & 15;
  const int ks = (lane >> 4) * 8;
  const unsigned short* p = base + n * stride + k0 + ks;
  bf16x8 lo = *(const bf16x8*)p;
  bf16x8 hi = *(const bf16x8*)(p + 16);
  bf16x16 b;
#pragma unroll
  for (int i = 0; i < 8; ++i) { b[i] = lo[i]; b[i + 8] = hi[i]; }
  return b;
}

// ---------------------------------------------------------------------------
// fp32 -> bf16 conversion (vectorized x4)
// ---------------------------------------------------------------------------
__global__ void cvt_f32_to_bf16(const float* __restrict__ in,
                                unsigned short* __restrict__ out, int n4) {
  int i = blockIdx.x * blockDim.x + threadIdx.x;
  if (i >= n4) return;
  float4 f = ((const float4*)in)[i];
  ushort4 o;
  o.x = f2bf(f.x); o.y = f2bf(f.y); o.z = f2bf(f.z); o.w = f2bf(f.w);
  ((ushort4*)out)[i] = o;
}

// ---------------------------------------------------------------------------
// Tiled GEMM + bias: out[M,N] = A[M,K](bf16) * B[K,N](bf16) + bias[N]
// ---------------------------------------------------------------------------
#define BM 128
#define BN 128
#define BK 32
#define LDKG (BK + 8)

template <bool OUT_BF16>
__global__ void gemm_bias_kernel(const unsigned short* __restrict__ A,
                                 const unsigned short* __restrict__ B,
                                 const float* __restrict__ bias,
                                 void* __restrict__ outp,
                                 int M, int N, int K) {
  extern __shared__ unsigned short lds[];
  unsigned short* As = lds;                  // [BM][LDKG] row-major (MxK)
  unsigned short* Bs = lds + BM * LDKG;      // [BN][LDKG] N-major

  const int tid  = threadIdx.x;
  const int lane = tid & 31;
  const int wave = tid >> 5;
  const int wm   = wave >> 1;
  const int wn   = wave & 1;

  const int m0 = blockIdx.y * BM;
  const int n0 = blockIdx.x * BN;

  f32x8 acc[2][4];
#pragma unroll
  for (int mi = 0; mi < 2; ++mi)
#pragma unroll
    for (int ni = 0; ni < 4; ++ni) acc[mi][ni] = zero8();

  const int arow   = tid >> 1;
  const int apart  = tid & 1;
  const int brow   = tid >> 3;
  const int bchunk = tid & 7;

  for (int k0 = 0; k0 < K; k0 += BK) {
    {
      const unsigned short* g = A + (size_t)(m0 + arow) * K + k0 + apart * 16;
      uint4 p0 = *(const uint4*)g;
      uint4 p1 = *(const uint4*)(g + 8);
      *(uint4*)(As + arow * LDKG + apart * 16)     = p0;
      *(uint4*)(As + arow * LDKG + apart * 16 + 8) = p1;
    }
    {
      const unsigned short* g = B + (size_t)(k0 + brow) * N + n0 + bchunk * 16;
      uint4 p0 = *(const uint4*)g;
      uint4 p1 = *(const uint4*)(g + 8);
      unsigned short v[16];
      *(uint4*)(v)     = p0;
      *(uint4*)(v + 8) = p1;
#pragma unroll
      for (int j = 0; j < 16; ++j)
        Bs[(bchunk * 16 + j) * LDKG + brow] = v[j];
    }
    if (k0 + BK < K) {  // -> global_prefetch_b8
      __builtin_prefetch(A + (size_t)(m0 + arow) * K + k0 + BK + apart * 16, 0, 1);
      __builtin_prefetch(B + (size_t)(k0 + BK + brow) * N + n0 + bchunk * 16, 0, 1);
    }
    __syncthreads();

    bf16x16 afrag[2], bfrag[4];
#pragma unroll
    for (int mi = 0; mi < 2; ++mi)
      afrag[mi] = load_frag_A(As + (wm * 32 + mi * 16) * LDKG, LDKG, 0, lane);
#pragma unroll
    for (int ni = 0; ni < 4; ++ni)
      bfrag[ni] = load_frag_B(Bs + (wn * 64 + ni * 16) * LDKG, LDKG, 0, lane);
#pragma unroll
    for (int mi = 0; mi < 2; ++mi)
#pragma unroll
      for (int ni = 0; ni < 4; ++ni)
        acc[mi][ni] = wmma_bf16(afrag[mi], bfrag[ni], acc[mi][ni]);
    __syncthreads();
  }

  const int nl  = lane & 15;
  const int rhi = (lane >> 4) * 8;
#pragma unroll
  for (int mi = 0; mi < 2; ++mi) {
#pragma unroll
    for (int ni = 0; ni < 4; ++ni) {
      const int n  = n0 + wn * 64 + ni * 16 + nl;
      const float bv = bias[n];
#pragma unroll
      for (int j = 0; j < 8; ++j) {
        const int m = m0 + wm * 32 + mi * 16 + j + rhi;
        const float v = acc[mi][ni][j] + bv;
        if (OUT_BF16) ((unsigned short*)outp)[(size_t)m * N + n] = f2bf(v);
        else          ((float*)outp)[(size_t)m * N + n] = v;
      }
    }
  }
}

// ---------------------------------------------------------------------------
// Block-local attention: one workgroup per (b, h, block); 256 threads = 8 waves
// LDS: QK[256][136] (Q cols 0-63, K cols 64-127), V^T[64][264], P[256][264]
// ---------------------------------------------------------------------------
#define BS    256
#define HD    64
#define QKSTR 136   // 128 data halves + 8 pad (TDM pad: 4 dwords per 64 dwords)
#define VSTR  264
#define PSTR  264
#define QK_HALVES (BS * QKSTR)
#define VT_HALVES (HD * VSTR)

__global__ void attn_kernel(const unsigned short* __restrict__ qkv,
                            unsigned short* __restrict__ o) {
  extern __shared__ unsigned short lds[];
  unsigned short* QK = lds;                  // [256][QKSTR]
  unsigned short* VT = QK + QK_HALVES;       // [HD][VSTR]
  unsigned short* Ps = VT + VT_HALVES;       // [256][PSTR]

  const int tid  = threadIdx.x;
  const int lane = tid & 31;
  const int wave = tid >> 5;

  const int idx = blockIdx.x;                // b*256 + h*16 + blk
  const int b   = idx >> 8;
  const int h   = (idx >> 4) & 15;
  const int blk = idx & 15;
  const int m0  = b * 4096 + blk * 256;

  const unsigned short* grow = qkv + (size_t)(m0 + tid) * 3072 + h * 192;

#if HAVE_TDM
  // ---- TDM: one 256x128-element 2D tile stages Q|K with LDS padding ----
  if (wave == 0) {
    unsigned long long ga =
        (unsigned long long)(const unsigned short*)(qkv + (size_t)m0 * 3072 + h * 192);
    u32x4 g0;
    g0[0] = 1u;                                    // count=1, user descriptor
    g0[1] = 0u;                                    // lds_addr = 0 (QK at base)
    g0[2] = (unsigned)(ga & 0xFFFFFFFFull);        // global_addr[31:0]
    g0[3] = (unsigned)((ga >> 32) & 0x1FFFFFFull)  // global_addr[56:32]
            | (2u << 30);                          // type = 2 (image)
    const unsigned td0 = 3072, td1 = 16384;        // tensor dims (elements)
    const unsigned tile0 = 128, tile1 = 256;       // tile dims
    i32x8 g1;
    g1[0] = (int)((1u << 16)     // data_size = 2 bytes
                | (1u << 20)     // pad_enable
                | (5u << 22)     // pad_interval = 64 dwords (128 halves)
                | (3u << 25));   // pad_amount = 4 dwords (8 halves)
    g1[1] = (int)((td0 & 0xFFFFu) << 16);
    g1[2] = (int)((td0 >> 16) | ((td1 & 0xFFFFu) << 16));
    g1[3] = (int)((td1 >> 16) | (tile0 << 16));
    g1[4] = (int)tile1;                            // tile_dim1 | tile_dim2=0
    g1[5] = (int)3072;                             // tensor_dim0_stride lo
    g1[6] = 0;
    g1[7] = 0;
    i32x4 z4 = {0, 0, 0, 0};
#if defined(__clang_major__) && __clang_major__ >= 23
    i32x8 z8 = {0, 0, 0, 0, 0, 0, 0, 0};
    __builtin_amdgcn_tensor_load_to_lds(g0, g1, z4, z4, z8, 0);
#else
    __builtin_amdgcn_tensor_load_to_lds(g0, g1, z4, z4, 0);
#endif
  }
#else
  // ---- manual Q|K staging ----
  {
#pragma unroll
    for (int c8 = 0; c8 < 16; ++c8)
      *(uint4*)(QK + tid * QKSTR + c8 * 8) = *(const uint4*)(grow + c8 * 8);
  }
#endif

  // ---- V staged transposed (TDM cannot transpose) ----
  {
    unsigned short vv[64];
#pragma unroll
    for (int c8 = 0; c8 < 8; ++c8)
      *(uint4*)(vv + c8 * 8) = *(const uint4*)(grow + 128 + c8 * 8);
#pragma unroll
    for (int c = 0; c < 64; ++c)
      VT[c * VSTR + tid] = vv[c];
  }

#if HAVE_TDM
  if (wave == 0) __builtin_amdgcn_s_wait_tensorcnt(0);
#endif
  __syncthreads();

  const float scale = 0.125f;                // 1/sqrt(64)
  const int nl  = lane & 15;
  const int rhi = (lane >> 4) * 8;

  // ---- S = Q K^T * scale ; two-pass online softmax -> normalized P ----
#pragma unroll 1
  for (int si = 0; si < 2; ++si) {
    const int sr0 = wave * 32 + si * 16;
    bf16x16 qf0 = load_frag_A(QK + sr0 * QKSTR, QKSTR, 0,  lane);
    bf16x16 qf1 = load_frag_A(QK + sr0 * QKSTR, QKSTR, 32, lane);

    float mh[2][8], sh[2][8];

#pragma unroll
    for (int half = 0; half < 2; ++half) {
      f32x8 acc[8];
#pragma unroll
      for (int t = 0; t < 8; ++t) acc[t] = zero8();
#pragma unroll
      for (int t = 0; t < 8; ++t) {
        const int tt = half * 8 + t;
        bf16x16 kf0 = load_frag_B(QK + (tt * 16) * QKSTR + 64, QKSTR, 0,  lane);
        bf16x16 kf1 = load_frag_B(QK + (tt * 16) * QKSTR + 64, QKSTR, 32, lane);
        acc[t] = wmma_bf16(qf0, kf0, acc[t]);
        acc[t] = wmma_bf16(qf1, kf1, acc[t]);
      }
#pragma unroll
      for (int j = 0; j < 8; ++j) {
        float mx = -3.0e38f;
#pragma unroll
        for (int t = 0; t < 8; ++t) {
          acc[t][j] *= scale;
          mx = fmaxf(mx, acc[t][j]);
        }
#pragma unroll
        for (int off = 1; off < 16; off <<= 1)
          mx = fmaxf(mx, __shfl_xor(mx, off, 32));
        float sum = 0.0f;
        const int row = sr0 + j + rhi;
#pragma unroll
        for (int t = 0; t < 8; ++t) {
          float e = __expf(acc[t][j] - mx);
          sum += e;
          Ps[row * PSTR + (half * 8 + t) * 16 + nl] = f2bf(e);
        }
#pragma unroll
        for (int off = 1; off < 16; off <<= 1)
          sum += __shfl_xor(sum, off, 32);
        mh[half][j] = mx;
        sh[half][j] = sum;
      }
      sched_fence();   // keep the two halves' live ranges disjoint (no spills)
    }

    // merge: P_half *= exp(m_half - m) / total
#pragma unroll
    for (int j = 0; j < 8; ++j) {
      const float m  = fmaxf(mh[0][j], mh[1][j]);
      const float a0 = __expf(mh[0][j] - m);
      const float a1 = __expf(mh[1][j] - m);
      const float inv = 1.0f / (sh[0][j] * a0 + sh[1][j] * a1);
      const float c0 = a0 * inv, c1 = a1 * inv;
      const int row = sr0 + j + rhi;
#pragma unroll
      for (int t = 0; t < 8; ++t) {
        unsigned short* p0 = &Ps[row * PSTR + t * 16 + nl];
        unsigned short* p1 = p0 + 128;
        *p0 = f2bf(bf2f(*p0) * c0);
        *p1 = f2bf(bf2f(*p1) * c1);
      }
    }
  }
  __syncthreads();

  // ---- O = P V ----
#pragma unroll 1
  for (int si = 0; si < 2; ++si) {
    const int sr0 = wave * 32 + si * 16;
    f32x8 acc[4];
#pragma unroll
    for (int ni = 0; ni < 4; ++ni) acc[ni] = zero8();
#pragma unroll
    for (int kt = 0; kt < 8; ++kt) {
      bf16x16 pf = load_frag_A(Ps + sr0 * PSTR, PSTR, kt * 32, lane);
#pragma unroll
      for (int ni = 0; ni < 4; ++ni) {
        bf16x16 vf = load_frag_B(VT + (ni * 16) * VSTR, VSTR, kt * 32, lane);
        acc[ni] = wmma_bf16(pf, vf, acc[ni]);
      }
    }
#pragma unroll
    for (int ni = 0; ni < 4; ++ni)
#pragma unroll
      for (int j = 0; j < 8; ++j) {
        const int m = m0 + sr0 + j + rhi;
        const int c = h * 64 + ni * 16 + nl;
        o[(size_t)m * 1024 + c] = f2bf(acc[ni][j]);
      }
  }
}

// ---------------------------------------------------------------------------
// Host launcher
// ---------------------------------------------------------------------------
extern "C" void kernel_launch(void* const* d_in, const int* in_sizes, int n_in,
                              void* d_out, int out_size, void* d_ws, size_t ws_size,
                              hipStream_t stream) {
  (void)in_sizes; (void)n_in; (void)out_size; (void)ws_size;

  const float* x    = (const float*)d_in[0];
  const float* Wqkv = (const float*)d_in[1];
  const float* bqkv = (const float*)d_in[2];
  const float* Wout = (const float*)d_in[3];
  const float* bout = (const float*)d_in[4];

  const int Bb = 4, T = 4096, C = 1024;
  const int M  = Bb * T;        // 16384
  const int N1 = 3 * C;         // 3072

  unsigned short* xb    = (unsigned short*)d_ws;            // [M][C]    bf16
  unsigned short* wqkvb = xb    + (size_t)M * C;            // [C][3C]   bf16
  unsigned short* woutb = wqkvb + (size_t)C * N1;           // [C][C]    bf16
  unsigned short* qkvb  = woutb + (size_t)C * C;            // [M][3C]   bf16
  unsigned short* ob    = qkvb  + (size_t)M * N1;           // [M][C]    bf16

  cvt_f32_to_bf16<<<(M * C / 4 + 255) / 256, 256, 0, stream>>>(x, xb, M * C / 4);
  cvt_f32_to_bf16<<<(C * N1 / 4 + 255) / 256, 256, 0, stream>>>(Wqkv, wqkvb, C * N1 / 4);
  cvt_f32_to_bf16<<<(C * C / 4 + 255) / 256, 256, 0, stream>>>(Wout, woutb, C * C / 4);

  const size_t gemmLds = (size_t)(BM * LDKG + BN * LDKG) * sizeof(unsigned short);
  gemm_bias_kernel<true><<<dim3(N1 / BN, M / BM), 256, gemmLds, stream>>>(
      xb, wqkvb, bqkv, (void*)qkvb, M, N1, C);

  const size_t attnLds =
      (size_t)(QK_HALVES + VT_HALVES + BS * PSTR) * sizeof(unsigned short);
  attn_kernel<<<dim3(Bb * 16 * 16), 256, attnLds, stream>>>(qkvb, ob);

  gemm_bias_kernel<false><<<dim3(C / BN, M / BM), 256, gemmLds, stream>>>(
      ob, woutb, bout, d_out, M, C, C);
}
